// Ddm_61692910240498
// MI455X (gfx1250) — matrix operand order repaired and checked by reading, not compiled, not run on hardware
//
#include <hip/hip_runtime.h>
#include <hip/hip_bf16.h>

// ---------------------------------------------------------------------------
// Bayesian multihead MLP forward for MI455X (gfx1250, wave32, WMMA + TDM).
//   1) materialize bf16 weights W = mu + exp(ls)*eps  (memory-bound, one pass)
//   2) 4x TDM-staged, double-buffered bf16 WMMA GEMM + bias + ReLU
//      (block tile 64x128, BK=32; Tensor Data Mover fills LDS while WMMAs run)
//   3) per-row task-selected head (sparse version of the dense-masked einsum)
// ---------------------------------------------------------------------------

typedef __attribute__((ext_vector_type(16))) __bf16 v16bf;
typedef __attribute__((ext_vector_type(8)))  float  v8f;
typedef __attribute__((ext_vector_type(4)))  unsigned int v4u;
typedef __attribute__((ext_vector_type(4)))  unsigned int u32x4;
typedef __attribute__((ext_vector_type(8)))  int i32x8;
typedef __attribute__((ext_vector_type(4)))  int i32x4;

#define IN_DIM 1024
#define HID    2048
#define OUTD   10
#define NTASK  10
#define NEXTRA 3
#define BATCH  4096

#if defined(__gfx1250__) && __has_builtin(__builtin_amdgcn_tensor_load_to_lds)
#define USE_TDM 1
#else
#define USE_TDM 0
#endif

// ---------------- elementwise materialization kernels ----------------------

__global__ void bayes_mat_bf16(const float* __restrict__ mu,
                               const float* __restrict__ ls,
                               const float* __restrict__ eps,
                               __bf16* __restrict__ w, long n) {
  long i = (long)blockIdx.x * blockDim.x + threadIdx.x;
  long stride = (long)gridDim.x * blockDim.x;
  for (; i < n; i += stride)
    w[i] = (__bf16)(mu[i] + __expf(ls[i]) * eps[i]);
}

__global__ void bayes_mat_f32(const float* __restrict__ mu,
                              const float* __restrict__ ls,
                              const float* __restrict__ eps,
                              float* __restrict__ w, long n) {
  long i = (long)blockIdx.x * blockDim.x + threadIdx.x;
  long stride = (long)gridDim.x * blockDim.x;
  for (; i < n; i += stride)
    w[i] = mu[i] + __expf(ls[i]) * eps[i];
}

__global__ void f32_to_bf16(const float* __restrict__ x,
                            __bf16* __restrict__ y, long n) {
  long i = (long)blockIdx.x * blockDim.x + threadIdx.x;
  long stride = (long)gridDim.x * blockDim.x;
  for (; i < n; i += stride)
    y[i] = (__bf16)x[i];
}

// ---------------- TDM 2D tile load (ISA ch.8: Tensor DMA Descriptor) -------
// Loads a (rows x rowElems) bf16 tile (row stride = strideElems) from global
// into LDS at byte offset lds_off, row-major packed (rowElems*2 bytes/row).
#if USE_TDM
__device__ __forceinline__ void tdm_load_2d_bf16(unsigned lds_off,
                                                 const void* gtile,
                                                 unsigned rows,
                                                 unsigned rowElems,
                                                 unsigned tensorD0,      // elems
                                                 unsigned tensorD1,      // rows
                                                 unsigned long long strideElems) {
  const unsigned long long ga = (unsigned long long)gtile;
  u32x4 g0;
  g0[0] = 1u;                                      // count=1 (valid), user mode
  g0[1] = lds_off;                                 // lds_addr (bytes)
  g0[2] = (unsigned)(ga & 0xFFFFFFFFu);            // global_addr[31:0]
  g0[3] = (unsigned)((ga >> 32) & 0x01FFFFFFu)     // global_addr[56:32]
        | (2u << 30);                              // type=2 ("image")
  i32x8 g1;
  g1[0] = (int)(1u << 16);                         // wg_mask=0, data_size=1 (2B)
  g1[1] = (int)((tensorD0 & 0xFFFFu) << 16);       // tensor_dim0[15:0] @63:48
  g1[2] = (int)(((tensorD0 >> 16) & 0xFFFFu)       // tensor_dim0[31:16]
        | ((tensorD1 & 0xFFFFu) << 16));           // tensor_dim1[15:0]
  g1[3] = (int)(((tensorD1 >> 16) & 0xFFFFu)       // tensor_dim1[31:16]
        | ((rowElems & 0xFFFFu) << 16));           // tile_dim0
  g1[4] = (int)(rows & 0xFFFFu);                   // tile_dim1 (tile_dim2=0)
  g1[5] = (int)(unsigned)(strideElems & 0xFFFFFFFFu);       // dim0_stride[31:0]
  g1[6] = (int)(unsigned)((strideElems >> 32) & 0xFFFFu);   // dim0_stride[47:32]
  g1[7] = 0;                                       // tensor_dim1_stride=0 (2D)
  const i32x4 z4 = {0, 0, 0, 0};
#if __clang_major__ >= 23
  const i32x8 z8 = {0, 0, 0, 0, 0, 0, 0, 0};
  __builtin_amdgcn_tensor_load_to_lds(g0, g1, z4, z4, z8, 0);
#else
  __builtin_amdgcn_tensor_load_to_lds(g0, g1, z4, z4, 0);
#endif
}
#endif  // USE_TDM

// ---------------- fused GEMM + bias + ReLU ---------------------------------
// out[M,N] = act(A[M,K] @ W[N,K]^T + bias), bf16 in/out, f32 accumulate.
// 256 threads = 8 waves as 2(M) x 4(N); wave tile 32x32 (2x2 WMMA tiles).
// Block tile 64x128, BK=32, double-buffered LDS. Wave 0 drives the Tensor
// Data Mover: tile t+1 streams into LDS[buf^1] while all waves run WMMAs on
// LDS[buf]; completion is s_wait_tensorcnt + workgroup barrier.
// Fragment layouts per ISA 7.12.2 (wave32):
//   A 16x32 bf16: lane (half=l/16, m=l%16): row m, K=half*8+{0..7}, 16+half*8+{0..7}
//   B 32x16 bf16: lane holds col n=l%16, K=half*8+{0..7}, 16+half*8+{0..7}
//   C/D 16x16 f32: VGPR i = row i+8*half, col = l%16.
__global__ void __launch_bounds__(256)
gemm_bias_relu_bf16(const __bf16* __restrict__ A, const __bf16* __restrict__ W,
                    const float* __restrict__ bias, __bf16* __restrict__ out,
                    int M, int N, int K, int do_relu) {
  __shared__ __align__(16) __bf16 lsA[2][64 * 32];   // 8 KB
  __shared__ __align__(16) __bf16 lsB[2][128 * 32];  // 16 KB

  const int tid  = threadIdx.x;
  const int lane = tid & 31;
  const int wave = tid >> 5;
  const int wm   = wave & 1;   // 0..1 along M (32 rows each)
  const int wn   = wave >> 1;  // 0..3 along N (32 cols each)
  const int half = lane >> 4;
  const int lr   = lane & 15;

  const int blockM = blockIdx.y * 64;
  const int blockN = blockIdx.x * 128;

  v8f acc[2][2] = {{v8f{}, v8f{}}, {v8f{}, v8f{}}};
  const int tiles = K >> 5;

#if USE_TDM
  const __bf16* gAbase = A + (size_t)blockM * K;
  const __bf16* gBbase = W + (size_t)blockN * K;
  // prologue: DMA tile 0 into LDS buffer 0 (one wave issues; EXEC ignored)
  if (wave == 0) {
    tdm_load_2d_bf16((unsigned)(unsigned long long)(uintptr_t)&lsA[0][0],
                     gAbase, 64, 32, (unsigned)K, (unsigned)(M - blockM),
                     (unsigned long long)K);
    tdm_load_2d_bf16((unsigned)(unsigned long long)(uintptr_t)&lsB[0][0],
                     gBbase, 128, 32, (unsigned)K, (unsigned)(N - blockN),
                     (unsigned long long)K);
    __builtin_amdgcn_s_wait_tensorcnt(0);
  }
  __syncthreads();
#else
  // fallback: register-staged global->LDS pipeline
  const int arow = tid >> 2, aseg = tid & 3;
  const int brow0 = tid >> 2, brow1 = (tid >> 2) + 64, bseg = tid & 3;
  const __bf16* gA  = A + (size_t)(blockM + arow) * K + aseg * 8;
  const __bf16* gB0 = W + (size_t)(blockN + brow0) * K + bseg * 8;
  const __bf16* gB1 = W + (size_t)(blockN + brow1) * K + bseg * 8;
  {
    v4u ra  = *reinterpret_cast<const v4u*>(gA);
    v4u rb0 = *reinterpret_cast<const v4u*>(gB0);
    v4u rb1 = *reinterpret_cast<const v4u*>(gB1);
    *reinterpret_cast<v4u*>(&lsA[0][arow * 32 + aseg * 8])  = ra;
    *reinterpret_cast<v4u*>(&lsB[0][brow0 * 32 + bseg * 8]) = rb0;
    *reinterpret_cast<v4u*>(&lsB[0][brow1 * 32 + bseg * 8]) = rb1;
  }
  __syncthreads();
#endif

  for (int t = 0; t < tiles; ++t) {
    const int buf = t & 1;
    const bool more = (t + 1) < tiles;

#if USE_TDM
    // kick off DMA for tile t+1 into LDS[buf^1] (readers of buf^1 finished
    // before the barrier that admitted us into iteration t)
    if (more && wave == 0) {
      const int k0 = (t + 1) << 5;
      tdm_load_2d_bf16((unsigned)(unsigned long long)(uintptr_t)&lsA[buf ^ 1][0],
                       gAbase + k0, 64, 32, (unsigned)K, (unsigned)(M - blockM),
                       (unsigned long long)K);
      tdm_load_2d_bf16((unsigned)(unsigned long long)(uintptr_t)&lsB[buf ^ 1][0],
                       gBbase + k0, 128, 32, (unsigned)K, (unsigned)(N - blockN),
                       (unsigned long long)K);
    }
#else
    v4u na{}, nb0{}, nb1{};
    if (more) {
      const int k0 = (t + 1) << 5;
      na  = *reinterpret_cast<const v4u*>(gA + k0);
      nb0 = *reinterpret_cast<const v4u*>(gB0 + k0);
      nb1 = *reinterpret_cast<const v4u*>(gB1 + k0);
    }
#endif

    // fragment loads from LDS[buf]
    union F { v4u u[2]; v16bf v; };
    F af[2], bfr[2];
#pragma unroll
    for (int i = 0; i < 2; ++i) {
      const __bf16* pa = &lsA[buf][(wm * 32 + i * 16 + lr) * 32 + half * 8];
      af[i].u[0] = *reinterpret_cast<const v4u*>(pa);
      af[i].u[1] = *reinterpret_cast<const v4u*>(pa + 16);
      const __bf16* pb = &lsB[buf][(wn * 32 + i * 16 + lr) * 32 + half * 8];
      bfr[i].u[0] = *reinterpret_cast<const v4u*>(pb);
      bfr[i].u[1] = *reinterpret_cast<const v4u*>(pb + 16);
    }

#pragma unroll
    for (int i = 0; i < 2; ++i)
#pragma unroll
      for (int j = 0; j < 2; ++j)
        acc[i][j] = __builtin_amdgcn_wmma_f32_16x16x32_bf16(
            /*neg_a=*/false, af[i].v, /*neg_b=*/false, bfr[j].v,
            /*c_mod=*/(short)0, acc[i][j], /*reuse_a=*/false, /*reuse_b=*/false);

#if USE_TDM
    if (more && wave == 0)
      __builtin_amdgcn_s_wait_tensorcnt(0);  // tile t+1 landed in LDS[buf^1]
#else
    if (more) {
      *reinterpret_cast<v4u*>(&lsA[buf ^ 1][arow * 32 + aseg * 8])  = na;
      *reinterpret_cast<v4u*>(&lsB[buf ^ 1][brow0 * 32 + bseg * 8]) = nb0;
      *reinterpret_cast<v4u*>(&lsB[buf ^ 1][brow1 * 32 + bseg * 8]) = nb1;
    }
#endif
    __syncthreads();
  }

  // epilogue: bias + ReLU + bf16 store
#pragma unroll
  for (int i = 0; i < 2; ++i) {
#pragma unroll
    for (int j = 0; j < 2; ++j) {
      const int col = blockN + wn * 32 + j * 16 + lr;
      const float bv = bias[col];
#pragma unroll
      for (int r = 0; r < 8; ++r) {
        const int row = blockM + wm * 32 + i * 16 + r + 8 * half;
        float v = acc[i][j][r] + bv;
        if (do_relu) v = fmaxf(v, 0.0f);
        out[(size_t)row * N + col] = (__bf16)v;
      }
    }
  }
}

// ---------------- task-selected head ---------------------------------------
// out[b,o] = h[b,:] . HW[task[b], o, :] + HB[task[b], o]
__global__ void __launch_bounds__(256)
head_kernel(const __bf16* __restrict__ h, const __bf16* __restrict__ HW,
            const float* __restrict__ HB, const int* __restrict__ task,
            float* __restrict__ out) {
  const int idx = blockIdx.x * blockDim.x + threadIdx.x;
  if (idx >= BATCH * OUTD) return;
  const int b = idx / OUTD, o = idx % OUTD;
  const int t = task[b];
  const __bf16* hv = h + (size_t)b * HID;
  const __bf16* wv = HW + ((size_t)t * OUTD + o) * HID;
  float s = HB[t * OUTD + o];
#pragma unroll 8
  for (int k = 0; k < HID; ++k)
    s += (float)hv[k] * (float)wv[k];
  out[idx] = s;
}

// ---------------------------------------------------------------------------

extern "C" void kernel_launch(void* const* d_in, const int* in_sizes, int n_in,
                              void* d_out, int out_size, void* d_ws, size_t ws_size,
                              hipStream_t stream) {
  (void)in_sizes; (void)n_in; (void)out_size; (void)ws_size;

  const float* x      = (const float*)d_in[0];
  const float* mu_w0  = (const float*)d_in[1];
  const float* ls_w0  = (const float*)d_in[2];
  const float* mu_b0  = (const float*)d_in[3];
  const float* ls_b0  = (const float*)d_in[4];
  const float* mu_w   = (const float*)d_in[5];
  const float* ls_w   = (const float*)d_in[6];
  const float* mu_b   = (const float*)d_in[7];
  const float* ls_b   = (const float*)d_in[8];
  const float* mu_hw  = (const float*)d_in[9];
  const float* ls_hw  = (const float*)d_in[10];
  const float* mu_hb  = (const float*)d_in[11];
  const float* ls_hb  = (const float*)d_in[12];
  const float* eps_w0 = (const float*)d_in[13];
  const float* eps_b0 = (const float*)d_in[14];
  const float* eps_w  = (const float*)d_in[15];
  const float* eps_b  = (const float*)d_in[16];
  const float* eps_hw = (const float*)d_in[17];
  const float* eps_hb = (const float*)d_in[18];
  const int*   task   = (const int*)d_in[19];
  float* out = (float*)d_out;

  // workspace layout (256B-aligned slabs)
  char* base = (char*)d_ws;
  size_t off = 0;
  auto take = [&](size_t bytes) {
    char* p = base + off;
    off = (off + bytes + 255) & ~(size_t)255;
    return p;
  };
  __bf16* x_bf = (__bf16*)take((size_t)BATCH * IN_DIM * 2);
  __bf16* W0   = (__bf16*)take((size_t)HID * IN_DIM * 2);
  __bf16* W123 = (__bf16*)take((size_t)NEXTRA * HID * HID * 2);
  __bf16* WH   = (__bf16*)take((size_t)NTASK * OUTD * HID * 2);
  float*  b0   = (float*)take((size_t)HID * 4);
  float*  b123 = (float*)take((size_t)NEXTRA * HID * 4);
  float*  bh   = (float*)take((size_t)NTASK * OUTD * 4);
  __bf16* hA   = (__bf16*)take((size_t)BATCH * HID * 2);
  __bf16* hB   = (__bf16*)take((size_t)BATCH * HID * 2);

  // 1) convert x, materialize all weights/biases
  f32_to_bf16<<<dim3(2048), dim3(256), 0, stream>>>(x, x_bf,
                                                    (long)BATCH * IN_DIM);
  bayes_mat_bf16<<<dim3(2048), dim3(256), 0, stream>>>(mu_w0, ls_w0, eps_w0, W0,
                                                       (long)HID * IN_DIM);
  bayes_mat_bf16<<<dim3(4096), dim3(256), 0, stream>>>(mu_w, ls_w, eps_w, W123,
                                                       (long)NEXTRA * HID * HID);
  bayes_mat_bf16<<<dim3(256), dim3(256), 0, stream>>>(mu_hw, ls_hw, eps_hw, WH,
                                                      (long)NTASK * OUTD * HID);
  bayes_mat_f32<<<dim3(8), dim3(256), 0, stream>>>(mu_b0, ls_b0, eps_b0, b0,
                                                   (long)HID);
  bayes_mat_f32<<<dim3(24), dim3(256), 0, stream>>>(mu_b, ls_b, eps_b, b123,
                                                    (long)NEXTRA * HID);
  bayes_mat_f32<<<dim3(1), dim3(128), 0, stream>>>(mu_hb, ls_hb, eps_hb, bh,
                                                   (long)NTASK * OUTD);

  // 2) GEMM chain (grid: N/128 x M/64)
  dim3 blk(256);
  dim3 grd(HID / 128, BATCH / 64);
  gemm_bias_relu_bf16<<<grd, blk, 0, stream>>>(x_bf, W0, b0, hA,
                                               BATCH, HID, IN_DIM, 1);
  __bf16* src = hA; __bf16* dst = hB;
  for (int i = 0; i < NEXTRA; ++i) {
    gemm_bias_relu_bf16<<<grd, blk, 0, stream>>>(
        src, W123 + (size_t)i * HID * HID, b123 + (size_t)i * HID, dst,
        BATCH, HID, HID, 1);
    __bf16* tmp = src; src = dst; dst = tmp;
  }

  // 3) heads (final activations are in `src` after the last swap)
  head_kernel<<<dim3((BATCH * OUTD + 255) / 256), dim3(256), 0, stream>>>(
      src, WH, bh, task, out);
}
